// HumanModel_14542759264922
// MI455X (gfx1250) — matrix operand order repaired and checked by reading, not compiled, not run on hardware
//
#include <hip/hip_runtime.h>
#include <hip/hip_bf16.h>
#include <math.h>

#define V_    4
#define N_    65536
#define C_    128
#define H_    512
#define W_    512
#define MPOOL 1024
#define KNN_K 5
#define DH_   128
#define OUTC  (C_ + 3)     // 131 output channels per point
#define HW_   (H_ * W_)

typedef __attribute__((ext_vector_type(16))) _Float16 v16h;
typedef __attribute__((ext_vector_type(8)))  _Float16 v8h;
typedef __attribute__((ext_vector_type(8)))  float    v8f;

union A16 { v16h v; v8h h[2]; };

// ---------------------------------------------------------------------------
// Kernel 1: project points into all views, bilinear-sample zbuffer, build mask
// ---------------------------------------------------------------------------
__device__ __forceinline__ float ztap(const float* __restrict__ img, int y, int x, float w) {
    bool ok = (y >= 0) && (y < H_) && (x >= 0) && (x < W_);
    int yc = y < 0 ? 0 : (y > H_ - 1 ? H_ - 1 : y);
    int xc = x < 0 ? 0 : (x > W_ - 1 ? W_ - 1 : x);
    return ok ? w * img[yc * W_ + xc] : 0.0f;
}

__global__ __launch_bounds__(256)
void project_kernel(const float* __restrict__ pts,
                    const float* __restrict__ extr,   // [V,3,4]
                    const float* __restrict__ intr,   // [V,3,3]
                    const float* __restrict__ zbuffer,// [V,1,H,W]
                    float* __restrict__ U, float* __restrict__ Vc,
                    float* __restrict__ Dep, float* __restrict__ Zraw,
                    float* __restrict__ Mask)
{
    int n = blockIdx.x * blockDim.x + threadIdx.x;
    if (n >= N_) return;
    float px = pts[n * 3 + 0], py = pts[n * 3 + 1], pz = pts[n * 3 + 2];
    float m = 0.0f;
#pragma unroll
    for (int v = 0; v < V_; ++v) {
        const float* E = extr + v * 12;
        float cx = E[0] * px + E[1] * py + E[2]  * pz + E[3];
        float cy = E[4] * px + E[5] * py + E[6]  * pz + E[7];
        float cz = E[8] * px + E[9] * py + E[10] * pz + E[11];
        const float* I = intr + v * 9;
        float qx = I[0] * cx + I[1] * cy + I[2] * cz;
        float qy = I[3] * cx + I[4] * cy + I[5] * cz;
        float qz = I[6] * cx + I[7] * cy + I[8] * cz;
        float inv = 1.0f / (qz + 1e-9f);
        float uu = qx * inv, vv = qy * inv;

        float u0f = floorf(uu), v0f = floorf(vv);
        float du = uu - u0f, dv = vv - v0f;
        int x0 = (int)u0f, y0 = (int)v0f;
        const float* Z = zbuffer + (size_t)v * HW_;
        float z = ztap(Z, y0,     x0,     (1.0f - du) * (1.0f - dv))
                + ztap(Z, y0,     x0 + 1, du          * (1.0f - dv))
                + ztap(Z, y0 + 1, x0,     (1.0f - du) * dv)
                + ztap(Z, y0 + 1, x0 + 1, du * dv);

        U[v * N_ + n]    = uu;
        Vc[v * N_ + n]   = vv;
        Dep[v * N_ + n]  = cz;
        Zraw[v * N_ + n] = z;
        if (z == 0.0f) m = 1.0f;
    }
    Mask[n] = m;
}

// ---------------------------------------------------------------------------
// Kernel 2: KNN (K=5) over the 1024-point candidate pool; replace zbuf for
// masked points with inverse-distance-weighted candidate zbuf values.
// ---------------------------------------------------------------------------
__global__ __launch_bounds__(256)
void knn_kernel(const float* __restrict__ pts,
                const float* __restrict__ Mask,
                const float* __restrict__ Zraw,
                float* __restrict__ Zfin)
{
    __shared__ float cxs[MPOOL], cys[MPOOL], czs[MPOOL];
    for (int i = threadIdx.x; i < MPOOL; i += blockDim.x) {
        float bad = Mask[i];
        // bad candidate -> +inf x coordinate -> d2 = inf -> disp = 0 (excluded)
        cxs[i] = (bad != 0.0f) ? __builtin_inff() : pts[i * 3 + 0];
        cys[i] = pts[i * 3 + 1];
        czs[i] = pts[i * 3 + 2];
    }
    __syncthreads();

    int n = blockIdx.x * blockDim.x + threadIdx.x;
    if (n >= N_) return;
    float mask = Mask[n];
    if (mask != 0.0f) {
        float px = pts[n * 3 + 0], py = pts[n * 3 + 1], pz = pts[n * 3 + 2];
        float bd[KNN_K]; int bi[KNN_K];
#pragma unroll
        for (int k = 0; k < KNN_K; ++k) { bd[k] = __builtin_inff(); bi[k] = 0; }
        for (int j = 0; j < MPOOL; ++j) {
            float dx = px - cxs[j], dy = py - cys[j], dz = pz - czs[j];
            float d2 = dx * dx + dy * dy + dz * dz;
            if (d2 < bd[KNN_K - 1]) {
                int k = KNN_K - 1;
                while (k > 0 && bd[k - 1] > d2) { bd[k] = bd[k - 1]; bi[k] = bi[k - 1]; --k; }
                bd[k] = d2; bi[k] = j;
            }
        }
        float disp[KNN_K], s = 0.0f;
#pragma unroll
        for (int k = 0; k < KNN_K; ++k) {
            float d = sqrtf(fmaxf(bd[k], 0.0f));
            disp[k] = 1.0f / (d + 1e-8f);     // inf distance -> 0 weight
            s += disp[k];
        }
        float inv = 1.0f / (s + 1e-12f);
#pragma unroll
        for (int v = 0; v < V_; ++v) {
            float rep = 0.0f;
#pragma unroll
            for (int k = 0; k < KNN_K; ++k) rep += Zraw[v * N_ + bi[k]] * disp[k];
            Zfin[v * N_ + n] = rep * inv;
        }
    } else {
#pragma unroll
        for (int v = 0; v < V_; ++v) Zfin[v * N_ + n] = Zraw[v * N_ + n];
    }
}

// ---------------------------------------------------------------------------
// Kernel 3: fused bilinear feature/rgb gather + WMMA MLP + view softmax +
// attention-weighted output. One block (256 thr = 8 waves) per 16-point tile.
//
// GEMM: M = 64 rows (4 views x 16 pts), K = 128 (feat only), N = 128.
// The vis channel (K=128 of the reference's 129-K GEMM) is a rank-1 update
// folded into the epilogue: h = relu(acc + b1 + vis*W1[128]).
//
// Operand packing uses contiguous k order (k = 16h + s) in A and B:
//   A (16-bit 16x32): lane reads K = kb+16h..+7 and kb+16h+8..+15 -> 2x b128 LDS
//   B (32x16):        lane reads 16 consecutive K of one column  -> 1x 32B LDS
// so W1 is staged TRANSPOSED (w1t[d][k]) in LDS.
// ---------------------------------------------------------------------------
__global__ __launch_bounds__(256)
void fuse_kernel(const float* __restrict__ featg,  // [V,C,H,W]
                 const float* __restrict__ imgg,   // [V,3,H,W]
                 const float* __restrict__ W1g,    // [129,128]
                 const float* __restrict__ b1g,    // [128]
                 const float* __restrict__ w2g,    // [128,1]
                 const float* __restrict__ U, const float* __restrict__ Vc,
                 const float* __restrict__ Dep, const float* __restrict__ Zfin,
                 float* __restrict__ outg)         // [N,131]
{
    __shared__ __align__(32) _Float16 xl[V_ * 16 * C_];   // 16 KB feat samples
    __shared__ __align__(32) _Float16 w1t[DH_ * C_];      // 32 KB W1^T; reused as h
    __shared__ float visl[V_ * 16];
    __shared__ float w1r128[DH_];                         // W1 row 128 (vis), f32
    __shared__ float tapw[V_ * 16][4];
    __shared__ int   tapo[V_ * 16][4];
    __shared__ float rgbl[V_ * 16 * 3];
    __shared__ float b1l[DH_], w2l[DH_];
    __shared__ float scorel[V_ * 16];
    __shared__ float attnl[V_ * 16];

    const int t  = threadIdx.x;
    const int n0 = blockIdx.x * 16;

    // ---- stage W1 transposed (f32 -> f16), b1, w2 into LDS ----
    for (int idx = t; idx < (C_ + 1) * DH_; idx += 256) {
        float val = W1g[idx];           // coalesced global read
        int k = idx >> 7, d = idx & (DH_ - 1);
        if (k < C_) w1t[d * C_ + k] = (_Float16)val;   // transposed LDS write
        else        w1r128[d] = val;                    // vis row kept f32
    }
    if (t < DH_) { b1l[t] = b1g[t]; w2l[t] = w2g[t]; }

    // ---- bilinear tap setup + vis per (view, point) ----
    if (t < V_ * 16) {
        int v = t >> 4, p = t & 15, n = n0 + p;
        float uu = U[v * N_ + n], vv = Vc[v * N_ + n];
        float u0f = floorf(uu), v0f = floorf(vv);
        float du = uu - u0f, dv = vv - v0f;
        int x0 = (int)u0f, y0 = (int)v0f;
        int xs[4] = { x0, x0 + 1, x0,     x0 + 1 };
        int ys[4] = { y0, y0,     y0 + 1, y0 + 1 };
        float ws[4] = { (1.0f - du) * (1.0f - dv), du * (1.0f - dv),
                        (1.0f - du) * dv,          du * dv };
#pragma unroll
        for (int k = 0; k < 4; ++k) {
            bool ok = (xs[k] >= 0) && (xs[k] < W_) && (ys[k] >= 0) && (ys[k] < H_);
            int xc = xs[k] < 0 ? 0 : (xs[k] > W_ - 1 ? W_ - 1 : xs[k]);
            int yc = ys[k] < 0 ? 0 : (ys[k] > H_ - 1 ? H_ - 1 : ys[k]);
            tapw[t][k] = ok ? ws[k] : 0.0f;
            tapo[t][k] = yc * W_ + xc;
        }
        float d = Dep[v * N_ + n], z = Zfin[v * N_ + n];
        visl[t] = 1.0f - fabsf(z - d) / (z + 1e-9f);
    }
    __syncthreads();

    // ---- gather 128 feature channels per (v,p): thread covers 32 channels ----
    {
        int vp = t & 63, cg = t >> 6;
        int v = vp >> 4;
        float w0 = tapw[vp][0], w1 = tapw[vp][1], w2 = tapw[vp][2], w3 = tapw[vp][3];
        int   o0 = tapo[vp][0], o1 = tapo[vp][1], o2 = tapo[vp][2], o3 = tapo[vp][3];
        const float* base = featg + (size_t)v * C_ * HW_;
        for (int c = cg * 32; c < cg * 32 + 32; ++c) {
            const float* f = base + (size_t)c * HW_;
            float val = w0 * f[o0] + w1 * f[o1] + w2 * f[o2] + w3 * f[o3];
            xl[vp * C_ + c] = (_Float16)val;
        }
        if (t < 192) {  // rgb gather: 64 (v,p) x 3 channels
            int vp2 = t % 64, cc = t / 64;
            int v2 = vp2 >> 4;
            const float* f = imgg + ((size_t)v2 * 3 + cc) * HW_;
            rgbl[vp2 * 3 + cc] = tapw[vp2][0] * f[tapo[vp2][0]]
                               + tapw[vp2][1] * f[tapo[vp2][1]]
                               + tapw[vp2][2] * f[tapo[vp2][2]]
                               + tapw[vp2][3] * f[tapo[vp2][3]];
        }
    }
    __syncthreads();

    // ---- WMMA GEMM: acc = feat @ W1[0:128]  (16 v_wmma per wave) ----
    const int wid  = t >> 5;
    const int lane = t & 31;
    const int view = wid >> 1;          // M-tile (16 points of this view)
    const int ng   = wid & 1;           // N-tile group (4 tiles of 16 columns)
    const int lm   = lane & 15;         // A row / BD column within tile
    const int hh   = lane >> 4;         // half-wave selector

    const _Float16* xrow = xl + (view * 16 + lm) * C_;
    v8f acc[4] = {};
#pragma unroll
    for (int kc = 0; kc < 4; ++kc) {
        const int kb = kc * 32;
        A16 a;
        a.h[0] = *(const v8h*)(xrow + kb + 16 * hh);       // K = kb+16h .. +7
        a.h[1] = *(const v8h*)(xrow + kb + 16 * hh + 8);   // K = kb+16h+8 .. +15
#pragma unroll
        for (int tt = 0; tt < 4; ++tt) {
            const int d0 = (ng * 4 + tt) * 16 + lm;        // this lane's column
            v16h b = *(const v16h*)(w1t + d0 * C_ + kb + 16 * hh);
            acc[tt] = __builtin_amdgcn_wmma_f32_16x16x32_f16(
                false, a.v, false, b, (short)0, acc[tt], false, false);
        }
    }
    __syncthreads();   // all waves done reading w1t; safe to reuse as h storage

    // ---- h = relu(acc + b1 + vis*W1[128]) -> LDS (reuse w1t region, f16) ----
    _Float16* hl = w1t;                 // [64][128]
#pragma unroll
    for (int tt = 0; tt < 4; ++tt) {
        const int d0 = (ng * 4 + tt) * 16 + lm;
        const float b1v = b1l[d0], wv = w1r128[d0];
#pragma unroll
        for (int r = 0; r < 8; ++r) {
            int p = r + 8 * hh;        // C/D row layout: M = vgpr + 8*half
            int row = view * 16 + p;
            float hv = acc[tt][r] + b1v + visl[row] * wv;
            hl[row * DH_ + d0] = (_Float16)fmaxf(hv, 0.0f);
        }
    }
    __syncthreads();

    // ---- attention scores (vectorized h . w2) + softmax over views ----
    if (t < V_ * 16) {
        float s = 0.0f;
#pragma unroll
        for (int d = 0; d < DH_; d += 8) {
            v8h hv = *(const v8h*)(hl + t * DH_ + d);
#pragma unroll
            for (int e = 0; e < 8; ++e) s += (float)hv[e] * w2l[d + e];
        }
        scorel[t] = s;
    }
    __syncthreads();
    if (t < 16) {
        float mx = scorel[t];
#pragma unroll
        for (int v = 1; v < V_; ++v) mx = fmaxf(mx, scorel[v * 16 + t]);
        float ssum = 0.0f, e[V_];
#pragma unroll
        for (int v = 0; v < V_; ++v) { e[v] = expf(scorel[v * 16 + t] - mx); ssum += e[v]; }
        float inv = 1.0f / ssum;
#pragma unroll
        for (int v = 0; v < V_; ++v) attnl[v * 16 + t] = e[v] * inv;
    }
    __syncthreads();

    // ---- weighted output: [16 points][131 channels] ----
    for (int idx = t; idx < 16 * OUTC; idx += 256) {
        int p = idx / OUTC, c = idx % OUTC;
        float val = 0.0f;
        if (c < C_) {
#pragma unroll
            for (int v = 0; v < V_; ++v)
                val += attnl[v * 16 + p] * (float)xl[(v * 16 + p) * C_ + c];
        } else {
            int cc = c - C_;
#pragma unroll
            for (int v = 0; v < V_; ++v)
                val += attnl[v * 16 + p] * rgbl[(v * 16 + p) * 3 + cc];
        }
        outg[(size_t)(n0 + p) * OUTC + c] = val;
    }
}

// ---------------------------------------------------------------------------
extern "C" void kernel_launch(void* const* d_in, const int* in_sizes, int n_in,
                              void* d_out, int out_size, void* d_ws, size_t ws_size,
                              hipStream_t stream) {
    const float* pts     = (const float*)d_in[0];  // [N,3]
    const float* extr    = (const float*)d_in[1];  // [V,3,4]
    const float* intr    = (const float*)d_in[2];  // [V,3,3]
    const float* zbuffer = (const float*)d_in[3];  // [V,1,H,W]
    const float* featg   = (const float*)d_in[4];  // [V,C,H,W]
    const float* imgg    = (const float*)d_in[5];  // [V,3,H,W]
    const float* W1g     = (const float*)d_in[6];  // [129,128]
    const float* b1g     = (const float*)d_in[7];  // [128]
    const float* w2g     = (const float*)d_in[8];  // [128,1]
    float* out           = (float*)d_out;          // [N,131]

    float* U    = (float*)d_ws;          // [V,N]
    float* Vc   = U    + (size_t)V_ * N_;
    float* Dep  = Vc   + (size_t)V_ * N_;
    float* Zraw = Dep  + (size_t)V_ * N_;
    float* Zfin = Zraw + (size_t)V_ * N_;
    float* Mask = Zfin + (size_t)V_ * N_;  // [N]

    project_kernel<<<N_ / 256, 256, 0, stream>>>(pts, extr, intr, zbuffer,
                                                 U, Vc, Dep, Zraw, Mask);
    knn_kernel<<<N_ / 256, 256, 0, stream>>>(pts, Mask, Zraw, Zfin);
    fuse_kernel<<<N_ / 16, 256, 0, stream>>>(featg, imgg, W1g, b1g, w2g,
                                             U, Vc, Dep, Zfin, out);
    (void)in_sizes; (void)n_in; (void)out_size; (void)ws_size;
}
